// MNISTnet_17583596110467
// MI455X (gfx1250) — compile-verified, matrix operand order checked by reading, not run on hardware
//
#include <hip/hip_runtime.h>

typedef __attribute__((ext_vector_type(16))) _Float16 v16h;
typedef __attribute__((ext_vector_type(8)))  float    v8f;

#define T_STEPS 10
#define BATCH   32
#define CIN_CH  3
#define COUT_CH 64
#define HDIM    64
#define WDIM    64
#define HW      4096          // 64*64
#define KRED    27            // CIN*3*3, padded to 32 inside one WMMA
#define SXW     68            // padded row width: slots for w = -1 .. 66
#define DECAY   0.2f
#define INH     1.625f

// -------- workspace layout (bytes) --------
// [0,    4096)  : A fragments, half, 4 Mtiles * 32 lanes * 16 halves
// [4096, 6656)  : thr_enc, uint, T*COUT = 640 entries (monotone-encoded max)

// Monotone order-preserving float<->uint encoding so atomicMax works on fp32.
__device__ __forceinline__ unsigned enc_f32(float f) {
    unsigned u = __float_as_uint(f);
    return (u & 0x80000000u) ? ~u : (u | 0x80000000u);
}
__device__ __forceinline__ float dec_f32(unsigned e) {
    unsigned u = (e & 0x80000000u) ? (e ^ 0x80000000u) : ~e;
    return __uint_as_float(u);
}

// ISA 7.12.2 16-bit A-matrix 16x32 layout: within-lane half index j -> K.
// VGPR 0..3 hold K=0..7 (lanes 0-15) / K=8..15 (lanes 16-31),
// VGPR 4..7 hold K=16..23 / K=24..31. B (32x16) mirrors this per column.
__device__ __forceinline__ int kmap(int lane, int j) {
    int hi = (lane >> 4) & 1;
    int v = j >> 1, h = j & 1;
    int K = (v < 4) ? (2 * v + h) : (16 + 2 * (v - 4) + h);
    return K + 8 * hi;
}

// Per-lane LDS dword offsets into the staged rows for the 16 B-fragment
// halves. Time-invariant: computed ONCE before the t loop. K >= 27 lanes
// point at an always-zero padding slot (no branches at load time).
__device__ __forceinline__ void make_boffs(int* boffs, int lane, int row, int w) {
    #pragma unroll
    for (int j = 0; j < 16; ++j) {
        int K = kmap(lane, j);
        int off = 67;                     // sx[0][0][67]: always zero
        if (K < KRED) {
            int cin = K / 9;
            int dy1 = (K % 9) / 3;        // 0..2 == dy+1
            int dx  = (K % 3) - 1;
            off = (cin * 4 + dy1 + row) * SXW + (w + dx + 1);
        }
        boffs[j] = off;
    }
}

// Stage input rows h0-1..h0+2 (x 3 cin) for a block covering output rows
// h0, h0+1 of image b at step t. W-padded: slot s holds w = s-1, zeros at
// both ends and for out-of-range rows -> fragment loads need no bounds checks.
__device__ __forceinline__ void stage_rows(const float* __restrict__ x,
                                           float* __restrict__ sxf,
                                           int t, int b, int h0, int tid) {
    for (int idx = tid; idx < CIN_CH * 4 * SXW; idx += 256) {
        int cin = idx / (4 * SXW);
        int rem = idx - cin * (4 * SXW);
        int r   = rem / SXW;              // row slot 0..3 -> h0-1..h0+2
        int ws  = rem - r * SXW;
        int wx  = ws - 1;
        int hy  = h0 - 1 + r;
        float v = 0.0f;
        if (hy >= 0 && hy < HDIM && wx >= 0 && wx < WDIM)
            v = x[((size_t)(t * BATCH + b) * CIN_CH + cin) * HW + hy * WDIM + wx];
        sxf[idx] = v;
    }
}

// Kernel 0: pack conv weights (COUT x 27, fp32) into WMMA A-fragment layout
// as fp16 (K padded 27->32 with zeros), and init the threshold atomics.
__global__ void prep_kernel(const float* __restrict__ Wt,
                            _Float16* __restrict__ Afrag,
                            unsigned* __restrict__ thr_enc) {
    int tid = blockIdx.x * blockDim.x + threadIdx.x;
    if (tid < 4 * 32 * 16) {
        int j = tid & 15, lane = (tid >> 4) & 31, mt = tid >> 9;
        int M = mt * 16 + (lane & 15);
        int K = kmap(lane, j);
        float val = (K < KRED) ? Wt[M * KRED + K] : 0.0f;
        Afrag[tid] = (_Float16)val;
    } else if (tid < 4 * 32 * 16 + T_STEPS * COUT_CH) {
        thr_enc[tid - 4 * 32 * 16] = enc_f32(-3.4e38f);  // encoded -FLT_MAX
    }
}

// Kernel A: WMMA conv -> per-(t,channel) max (adaptive thresholds).
// Reduction: every lane ds_max_u32's its 32 (channel, position) products
// into a 64-entry encoded LDS array; after a barrier, 64 threads forward to
// global with one atomic each. No shuffles, no divergent atomic tails.
__global__ void __launch_bounds__(256) conv_thr_kernel(
    const float* __restrict__ x, const _Float16* __restrict__ Afrag,
    unsigned* __restrict__ thr_enc) {
    __shared__ float    sxf[CIN_CH * 4 * SXW];
    __shared__ unsigned smax[COUT_CH];
    const int tid  = threadIdx.x;
    const int wave = tid >> 5;
    const int lane = tid & 31;
    const int btile = blockIdx.x * 8;               // first tile of block
    const int b    = btile >> 8;                    // 256 tiles per image
    const int h0   = ((btile & 255) << 4) >> 6;     // block's first row (even)
    const int row  = wave >> 2;                     // 0 or 1
    const int w    = ((wave & 3) << 4) + (lane & 15);
    const int hi8  = (lane >> 4) << 3;

    int boffs[16];
    make_boffs(boffs, lane, row, w);
    v16h a0 = *(const v16h*)(Afrag + (0 * 32 + lane) * 16);
    v16h a1 = *(const v16h*)(Afrag + (1 * 32 + lane) * 16);
    v16h a2 = *(const v16h*)(Afrag + (2 * 32 + lane) * 16);
    v16h a3 = *(const v16h*)(Afrag + (3 * 32 + lane) * 16);

    for (int t = 0; t < T_STEPS; ++t) {
        __syncthreads();                 // previous iter's reads done
        stage_rows(x, sxf, t, b, h0, tid);
        if (tid < COUT_CH) smax[tid] = enc_f32(-3.4e38f);
        __syncthreads();

        v16h bfrag;
        #pragma unroll
        for (int j = 0; j < 16; ++j) bfrag[j] = (_Float16)sxf[boffs[j]];

        #pragma unroll
        for (int mt = 0; mt < 4; ++mt) {
            v16h afrag = (mt == 0) ? a0 : (mt == 1) ? a1 : (mt == 2) ? a2 : a3;
            v8f c = {};
            c = __builtin_amdgcn_wmma_f32_16x16x32_f16(
                    false, afrag, false, bfrag, (short)0, c, false, false);
            #pragma unroll
            for (int r = 0; r < 8; ++r)
                atomicMax(&smax[mt * 16 + r + hi8], enc_f32(c[r]));
        }

        __syncthreads();                 // all waves' ds atomics done
        if (tid < COUT_CH)
            atomicMax(&thr_enc[t * COUT_CH + tid], smax[tid]);
    }
}

// Kernel B: conv recomputed with WMMA, fused with LIF + ASF + WTA +
// inhibition. Membranes (32 channels/lane; 64 per position split between
// lanes n and n+16) live in registers across all 10 steps. Per-lane
// thresholds cached to registers each step via ds_load_b128. WTA uses one
// shfl_xor(16) exchange with argmax-first tie-breaking.
__global__ void __launch_bounds__(256) conv_lif_kernel(
    const float* __restrict__ x, const _Float16* __restrict__ Afrag,
    const unsigned* __restrict__ thr_enc, float* __restrict__ out) {
    __shared__ float sxf[CIN_CH * 4 * SXW];
    __shared__ float sthr[COUT_CH];
    const int tid  = threadIdx.x;
    const int wave = tid >> 5;
    const int lane = tid & 31;
    const int btile = blockIdx.x * 8;
    const int b    = btile >> 8;
    const int h0   = ((btile & 255) << 4) >> 6;
    const int row  = wave >> 2;
    const int w    = ((wave & 3) << 4) + (lane & 15);
    const int hi8  = (lane >> 4) << 3;
    const int hw   = (h0 + row) * WDIM + w;         // this lane's position

    int boffs[16];
    make_boffs(boffs, lane, row, w);
    v16h a0 = *(const v16h*)(Afrag + (0 * 32 + lane) * 16);
    v16h a1 = *(const v16h*)(Afrag + (1 * 32 + lane) * 16);
    v16h a2 = *(const v16h*)(Afrag + (2 * 32 + lane) * 16);
    v16h a3 = *(const v16h*)(Afrag + (3 * 32 + lane) * 16);

    float memv[4][8];
    #pragma unroll
    for (int mt = 0; mt < 4; ++mt)
        #pragma unroll
        for (int r = 0; r < 8; ++r) memv[mt][r] = 0.0f;

    for (int t = 0; t < T_STEPS; ++t) {
        __syncthreads();
        stage_rows(x, sxf, t, b, h0, tid);
        if (tid < COUT_CH)
            sthr[tid] = dec_f32(thr_enc[t * COUT_CH + tid]) + 1e-4f;
        __syncthreads();

        // cache this lane's 32 channel thresholds (2x b128 LDS loads per mt)
        float thrv[4][8];
        #pragma unroll
        for (int mt = 0; mt < 4; ++mt) {
            float4 qa = *(const float4*)&sthr[mt * 16 + hi8];
            float4 qb = *(const float4*)&sthr[mt * 16 + hi8 + 4];
            thrv[mt][0] = qa.x; thrv[mt][1] = qa.y;
            thrv[mt][2] = qa.z; thrv[mt][3] = qa.w;
            thrv[mt][4] = qb.x; thrv[mt][5] = qb.y;
            thrv[mt][6] = qb.z; thrv[mt][7] = qb.w;
        }

        v16h bfrag;
        #pragma unroll
        for (int j = 0; j < 16; ++j) bfrag[j] = (_Float16)sxf[boffs[j]];

        // integrate: conv (WMMA) -> ASF -> leaky membrane update
        #pragma unroll
        for (int mt = 0; mt < 4; ++mt) {
            v16h afrag = (mt == 0) ? a0 : (mt == 1) ? a1 : (mt == 2) ? a2 : a3;
            v8f c = {};
            c = __builtin_amdgcn_wmma_f32_16x16x32_f16(
                    false, afrag, false, bfrag, (short)0, c, false, false);
            #pragma unroll
            for (int r = 0; r < 8; ++r) {
                float thr = thrv[mt][r];
                float cur = fmaxf(c[r], 0.0f);
                float z   = (cur - 0.4f * thr) * (8.0f / thr);
                float asf = thr / (1.0f + __expf(-z));
                memv[mt][r] = memv[mt][r] * DECAY + asf;
            }
        }

        // lane-local argmax over spiking channels (scan is channel-ascending)
        float best = -3.4e38f;
        int   bch  = COUT_CH;       // COUT_CH == "no spike"
        #pragma unroll
        for (int mt = 0; mt < 4; ++mt)
            #pragma unroll
            for (int r = 0; r < 8; ++r) {
                float m = memv[mt][r];
                if (m > thrv[mt][r] && (bch == COUT_CH || m > best)) {
                    best = m; bch = mt * 16 + r + hi8;
                }
            }
        // combine with partner lane holding the other 32 channels
        float obest = __shfl_xor(best, 16, 32);
        int   obch  = __shfl_xor(bch, 16, 32);
        int winner = bch;
        if (obch != COUT_CH &&
            (bch == COUT_CH || obest > best || (obest == best && obch < bch)))
            winner = obch;
        const bool any_sp = (winner != COUT_CH);

        // emit one-hot spikes, reset winner, inhibit losers
        const size_t obase = (size_t)(t * BATCH + b) * COUT_CH * HW + hw;
        #pragma unroll
        for (int mt = 0; mt < 4; ++mt)
            #pragma unroll
            for (int r = 0; r < 8; ++r) {
                int ch = mt * 16 + r + hi8;
                bool s = (ch == winner);
                out[obase + (size_t)ch * HW] = s ? 1.0f : 0.0f;
                if (s)           memv[mt][r] = 0.0f;
                else if (any_sp) memv[mt][r] -= INH * thrv[mt][r];
            }
    }
}

extern "C" void kernel_launch(void* const* d_in, const int* in_sizes, int n_in,
                              void* d_out, int out_size, void* d_ws, size_t ws_size,
                              hipStream_t stream) {
    const float* x  = (const float*)d_in[0];  // (T,B,CIN,H,W) fp32 spikes
    const float* Wt = (const float*)d_in[1];  // (COUT,CIN,3,3) fp32
    char* ws = (char*)d_ws;
    _Float16* Afrag   = (_Float16*)ws;
    unsigned* thr_enc = (unsigned*)(ws + 4096);
    float* out = (float*)d_out;

    prep_kernel<<<16, 256, 0, stream>>>(Wt, Afrag, thr_enc);
    conv_thr_kernel<<<1024, 256, 0, stream>>>(x, Afrag, thr_enc);
    conv_lif_kernel<<<1024, 256, 0, stream>>>(x, Afrag, thr_enc, out);
}
